// CBTree_27504970564116
// MI455X (gfx1250) — matrix operand order repaired and checked by reading, not compiled, not run on hardware
//
#include <hip/hip_runtime.h>
#include <hip/hip_bf16.h>

// CBTree forward, MI455X (gfx1250), wave32 + v_wmma_f32_16x16x32_bf16.
//
// Per level l (l = 7..0, n = 4^l):
//   U[m,:] = 1*h[4m] + (2/3)*h[4m+1] + (1/3)*h[4m+2]
//   V[m,:] =           (1/3)*h[4m+1] + (2/3)*h[4m+2] + 1*h[4m+3]
//   h_new  = tanh(x_l + U @ wl.T + V @ wr.T)
// bf16 WMMA inputs, fp32 accumulation, bf16 intermediate h, fp32 final out.
//
// Hot levels (n % 32 == 0, i.e. l>=3) use a 2x2 register-blocked full-tile
// kernel: 4 independent accumulators, 1 fragment load per WMMA, no edge masks.
// Small levels (n = 16, 4, 1) use the masked generic 1x1 kernel.

typedef __attribute__((ext_vector_type(16))) __bf16 v16bf;
typedef __attribute__((ext_vector_type(8)))  __bf16 v8bf;
typedef __attribute__((ext_vector_type(8)))  float  v8f;

#define DD 512

// ---- weight fp32 -> bf16 (row-major [e][d], unchanged layout) ----
__global__ void cvt_w_kernel(const float* __restrict__ wl, const float* __restrict__ wr,
                             __bf16* __restrict__ wlb, __bf16* __restrict__ wrb) {
    int i = blockIdx.x * blockDim.x + threadIdx.x;
    if (i < DD * DD) {
        wlb[i] = (__bf16)wl[i];
        wrb[i] = (__bf16)wr[i];
    }
}

// ---- child combine: fp32 leaves -> bf16 U,V ----
__global__ void combine_leaf_kernel(const float* __restrict__ child,
                                    __bf16* __restrict__ U, __bf16* __restrict__ V, int n) {
    int i = blockIdx.x * blockDim.x + threadIdx.x;
    if (i >= n * DD) return;
    int m = i >> 9;
    int d = i & (DD - 1);
    const float* c = child + (size_t)(4 * m) * DD + d;
    float c0 = c[0], c1 = c[DD], c2 = c[2 * DD], c3 = c[3 * DD];
    float u = c0 + (2.f / 3.f) * c1 + (1.f / 3.f) * c2;
    float v =      (1.f / 3.f) * c1 + (2.f / 3.f) * c2 + c3;
    U[i] = (__bf16)u;
    V[i] = (__bf16)v;
}

// ---- child combine: bf16 h -> bf16 U,V ----
__global__ void combine_h_kernel(const __bf16* __restrict__ child,
                                 __bf16* __restrict__ U, __bf16* __restrict__ V, int n) {
    int i = blockIdx.x * blockDim.x + threadIdx.x;
    if (i >= n * DD) return;
    int m = i >> 9;
    int d = i & (DD - 1);
    const __bf16* c = child + (size_t)(4 * m) * DD + d;
    float c0 = (float)c[0], c1 = (float)c[DD], c2 = (float)c[2 * DD], c3 = (float)c[3 * DD];
    float u = c0 + (2.f / 3.f) * c1 + (1.f / 3.f) * c2;
    float v =      (1.f / 3.f) * c1 + (2.f / 3.f) * c2 + c3;
    U[i] = (__bf16)u;
    V[i] = (__bf16)v;
}

// A fragment (16-bit, 16x32): lane<16 -> K k..k+15,
//                             lane>=16 -> K k+16..k+31
__device__ __forceinline__ v16bf load_a_frag(const __bf16* row, int k, int half) {
    return *(const v16bf*)(row + k + half * 16);
}

// B fragment (16-bit, 32x16): lane<16 -> K k..k+15 of col e,
//                             lane>=16 -> K k+16..k+31 of col e
__device__ __forceinline__ v16bf load_b_frag(const __bf16* row, int k, int half) {
    return *(const v16bf*)(row + k + half * 16);
}

// ---- Hot-path GEMM + bias + tanh: 2x2 tiles per wave, n % 32 == 0. ----
// C[m,e] = tanh( X[m,e] + sum_k U[m,k]*WL[e,k] + V[m,k]*WR[e,k] )
__global__ void __launch_bounds__(256)
gemm_tanh_2x2_kernel(const __bf16* __restrict__ U, const __bf16* __restrict__ V,
                     const __bf16* __restrict__ WL, const __bf16* __restrict__ WR,
                     const float* __restrict__ X,
                     __bf16* __restrict__ hOut, float* __restrict__ fOut, int n) {
    const int lane = threadIdx.x & 31;
    const int wave = threadIdx.x >> 5;
    const int rowPairs = n >> 5;                    // tilesM / 2
    const int tid = blockIdx.x * 8 + wave;          // wave-uniform
    if (tid >= rowPairs * 16) return;               // wave-uniform exit
    const int rowPair = tid >> 4;                   // 16 column pairs (512 / 32)
    const int colPair = tid & 15;

    const int half = lane >> 4;
    const int lr   = lane & 15;

    const int row0 = rowPair * 32 + lr;             // A rows of the two row tiles
    const int row1 = row0 + 16;
    const int e0   = colPair * 32 + lr;             // B cols of the two col tiles
    const int e1   = e0 + 16;

    const __bf16* uRow0 = U + (size_t)row0 * DD;
    const __bf16* uRow1 = U + (size_t)row1 * DD;
    const __bf16* vRow0 = V + (size_t)row0 * DD;
    const __bf16* vRow1 = V + (size_t)row1 * DD;
    const __bf16* wlRow0 = WL + (size_t)e0 * DD;
    const __bf16* wlRow1 = WL + (size_t)e1 * DD;
    const __bf16* wrRow0 = WR + (size_t)e0 * DD;
    const __bf16* wrRow1 = WR + (size_t)e1 * DD;

    v8f acc00 = {}, acc01 = {}, acc10 = {}, acc11 = {};

#pragma unroll 4
    for (int k = 0; k < DD; k += 32) {
        v16bf aU0 = load_a_frag(uRow0, k, half);
        v16bf aU1 = load_a_frag(uRow1, k, half);
        v16bf aV0 = load_a_frag(vRow0, k, half);
        v16bf aV1 = load_a_frag(vRow1, k, half);
        v16bf bL0 = load_b_frag(wlRow0, k, half);
        v16bf bL1 = load_b_frag(wlRow1, k, half);
        v16bf bR0 = load_b_frag(wrRow0, k, half);
        v16bf bR1 = load_b_frag(wrRow1, k, half);

        acc00 = __builtin_amdgcn_wmma_f32_16x16x32_bf16(false, aU0, false, bL0, (short)0, acc00, false, false);
        acc01 = __builtin_amdgcn_wmma_f32_16x16x32_bf16(false, aU0, false, bL1, (short)0, acc01, false, false);
        acc10 = __builtin_amdgcn_wmma_f32_16x16x32_bf16(false, aU1, false, bL0, (short)0, acc10, false, false);
        acc11 = __builtin_amdgcn_wmma_f32_16x16x32_bf16(false, aU1, false, bL1, (short)0, acc11, false, false);
        acc00 = __builtin_amdgcn_wmma_f32_16x16x32_bf16(false, aV0, false, bR0, (short)0, acc00, false, false);
        acc01 = __builtin_amdgcn_wmma_f32_16x16x32_bf16(false, aV0, false, bR1, (short)0, acc01, false, false);
        acc10 = __builtin_amdgcn_wmma_f32_16x16x32_bf16(false, aV1, false, bR0, (short)0, acc10, false, false);
        acc11 = __builtin_amdgcn_wmma_f32_16x16x32_bf16(false, aV1, false, bR1, (short)0, acc11, false, false);
    }

    // C/D layout: lanes 0-15: N=lane, VGPR j -> M=j ; lanes 16-31: N=lane-16, VGPR j -> M=j+8
    const int mBase = rowPair * 32 + half * 8;
#pragma unroll
    for (int j = 0; j < 8; ++j) {
        const int m0 = mBase + j;
        const int m1 = m0 + 16;
        const float v00 = tanhf(X[(size_t)m0 * DD + e0] + acc00[j]);
        const float v01 = tanhf(X[(size_t)m0 * DD + e1] + acc01[j]);
        const float v10 = tanhf(X[(size_t)m1 * DD + e0] + acc10[j]);
        const float v11 = tanhf(X[(size_t)m1 * DD + e1] + acc11[j]);
        if (fOut != nullptr) {
            fOut[(size_t)m0 * DD + e0] = v00;
            fOut[(size_t)m0 * DD + e1] = v01;
            fOut[(size_t)m1 * DD + e0] = v10;
            fOut[(size_t)m1 * DD + e1] = v11;
        } else {
            hOut[(size_t)m0 * DD + e0] = (__bf16)v00;
            hOut[(size_t)m0 * DD + e1] = (__bf16)v01;
            hOut[(size_t)m1 * DD + e0] = (__bf16)v10;
            hOut[(size_t)m1 * DD + e1] = (__bf16)v11;
        }
    }
}

// ---- Generic masked GEMM + bias + tanh (small levels: n = 16, 4, 1). ----
__global__ void __launch_bounds__(256)
gemm_tanh_kernel(const __bf16* __restrict__ U, const __bf16* __restrict__ V,
                 const __bf16* __restrict__ WL, const __bf16* __restrict__ WR,
                 const float* __restrict__ X,
                 __bf16* __restrict__ hOut, float* __restrict__ fOut, int n) {
    const int lane = threadIdx.x & 31;
    const int wave = threadIdx.x >> 5;
    const int tilesM = (n + 15) >> 4;
    const int tid = blockIdx.x * 8 + wave;          // wave-uniform
    if (tid >= tilesM * 32) return;                 // wave-uniform exit
    const int tileM = tid >> 5;                     // 32 column tiles (512/16)
    const int tileE = tid & 31;

    const int half = lane >> 4;
    const int lr   = lane & 15;

    const int rowA    = tileM * 16 + lr;
    const bool validA = rowA < n;
    const int rowC    = validA ? rowA : (n - 1);    // clamp for safe loads
    const int e       = tileE * 16 + lr;

    const __bf16* uRow  = U  + (size_t)rowC * DD;
    const __bf16* vRow  = V  + (size_t)rowC * DD;
    const __bf16* wlRow = WL + (size_t)e * DD;
    const __bf16* wrRow = WR + (size_t)e * DD;

    v8f acc = {};
    const v16bf zfrag = {};

    for (int k = 0; k < DD; k += 32) {
        v16bf aU = load_a_frag(uRow, k, half);
        if (!validA) aU = zfrag;                    // divergent VALU only; reconverges pre-WMMA
        v16bf bL = load_b_frag(wlRow, k, half);
        acc = __builtin_amdgcn_wmma_f32_16x16x32_bf16(false, aU, false, bL,
                                                      (short)0, acc, false, false);
        v16bf aV = load_a_frag(vRow, k, half);
        if (!validA) aV = zfrag;
        v16bf bR = load_b_frag(wrRow, k, half);
        acc = __builtin_amdgcn_wmma_f32_16x16x32_bf16(false, aV, false, bR,
                                                      (short)0, acc, false, false);
    }

    const int col = tileE * 16 + lr;
#pragma unroll
    for (int j = 0; j < 8; ++j) {
        const int m = tileM * 16 + half * 8 + j;
        if (m < n) {
            const float val = tanhf(X[(size_t)m * DD + col] + acc[j]);
            if (fOut != nullptr) fOut[(size_t)m * DD + col] = val;
            else                 hOut[(size_t)m * DD + col] = (__bf16)val;
        }
    }
}

extern "C" void kernel_launch(void* const* d_in, const int* in_sizes, int n_in,
                              void* d_out, int out_size, void* d_ws, size_t ws_size,
                              hipStream_t stream) {
    (void)in_sizes; (void)n_in; (void)out_size; (void)ws_size;

    const float* vec = (const float*)d_in[0];   // [87381, 512]
    const float* wl  = (const float*)d_in[1];   // [512, 512]
    const float* wr  = (const float*)d_in[2];   // [512, 512]
    float* out = (float*)d_out;                 // [512] (h of root, n=1)

    // Level offsets for B=4, depth=8: (4^l - 1)/3
    static const int offs[9] = {0, 1, 5, 21, 85, 341, 1365, 5461, 21845};

    // Workspace layout (bytes):
    //  wlb:  512 KiB   wrb: 512 KiB
    //  U:    16 MiB    V:   16 MiB   (max n = 4^7 = 16384 rows)
    //  hA:   16 MiB    hB:  16 MiB   (ping/pong)
    char* ws = (char*)d_ws;
    __bf16* wlb = (__bf16*)(ws);
    __bf16* wrb = (__bf16*)(ws + (512ull << 10));
    __bf16* Ub  = (__bf16*)(ws + (1ull << 20));
    __bf16* Vb  = (__bf16*)(ws + (17ull << 20));
    __bf16* hA  = (__bf16*)(ws + (33ull << 20));
    __bf16* hB  = (__bf16*)(ws + (49ull << 20));

    cvt_w_kernel<<<(DD * DD + 255) / 256, 256, 0, stream>>>(wl, wr, wlb, wrb);

    const __bf16* hIn = nullptr;
    bool ping = true;
    for (int l = 7; l >= 0; --l) {
        const int n = 1 << (2 * l);               // 4^l
        // 1) combine children into U, V
        const int cThreads = n * DD;
        const int cBlocks = (cThreads + 255) / 256;
        if (l == 7) {
            combine_leaf_kernel<<<cBlocks, 256, 0, stream>>>(
                vec + (size_t)offs[8] * DD, Ub, Vb, n);
        } else {
            combine_h_kernel<<<cBlocks, 256, 0, stream>>>(hIn, Ub, Vb, n);
        }
        // 2) GEMM + bias + tanh
        __bf16* hOut = nullptr;
        float*  fOut = nullptr;
        if (l == 0) fOut = out;
        else        hOut = ping ? hA : hB;

        if (n % 32 == 0) {
            const int waves = (n >> 5) * 16;      // rowPairs * colPairs
            const int gBlocks = (waves + 7) / 8;
            gemm_tanh_2x2_kernel<<<gBlocks, 256, 0, stream>>>(
                Ub, Vb, wlb, wrb, vec + (size_t)offs[l] * DD, hOut, fOut, n);
        } else {
            const int tilesM = (n + 15) / 16;
            const int gBlocks = (tilesM * 32 + 7) / 8;
            gemm_tanh_kernel<<<gBlocks, 256, 0, stream>>>(
                Ub, Vb, wlb, wrb, vec + (size_t)offs[l] * DD, hOut, fOut, n);
        }

        hIn = hOut;
        ping = !ping;
    }
}
